// FeatureGrouper_16673063043167
// MI455X (gfx1250) — compile-verified, matrix operand order checked by reading, not run on hardware
//
#include <hip/hip_runtime.h>
#include <math.h>

// ---------------------------------------------------------------------------
// FeatureGrouper for MI455X (gfx1250): all GEMMs via v_wmma_f32_16x16x32_bf16
// (bf16 inputs, fp32 accumulation), tiny 5x5 softmax in fp32 VALU.
//
// Sizes: B=32, N=4096, INPUT_DIM=256, BASE=32, A=5, E=48, H=2, HD=24,
//        AGG_DIM=160, EXTRA_IN=96, EXTRA_OUT=272, out channels = 512.
// Row index j in [0, B*N*A) maps to (b = j/20480, a = (j%20480)/4096,
// m = j%4096); attention groups are 5 consecutive j rows; out-proj rows in j
// order are exactly the contiguous agg portion of the output.
//
// Fragment layout (ISA 7.12.2, 16-bit 16x32 A; B mirrored): for lane L,
// hs = (L>>4)&1, element p holds K = (p&7) + ((p>>3)<<4) + (hs<<3), i.e. two
// contiguous 8-half runs starting at kb = kslice*32 + hs*8 and kb+16.
// Runs are 16B-aligned -> build fragments with b128 loads.
// ---------------------------------------------------------------------------

typedef __attribute__((ext_vector_type(16))) __bf16 v16bf;
typedef __attribute__((ext_vector_type(8)))  float  v8f;

// Convert 8 consecutive f32 (16B-aligned) into fragment elements [e0..e0+7].
__device__ __forceinline__ void cvt_run(v16bf& r, const int e0, const float* p) {
    const float4 a = *reinterpret_cast<const float4*>(p);
    const float4 b = *reinterpret_cast<const float4*>(p + 4);
    r[e0 + 0] = (__bf16)a.x; r[e0 + 1] = (__bf16)a.y;
    r[e0 + 2] = (__bf16)a.z; r[e0 + 3] = (__bf16)a.w;
    r[e0 + 4] = (__bf16)b.x; r[e0 + 5] = (__bf16)b.y;
    r[e0 + 6] = (__bf16)b.z; r[e0 + 7] = (__bf16)b.w;
}

// Full fragment from global f32 (both 8-wide runs valid).
__device__ __forceinline__ v16bf gfrag(const float* base, const int kb) {
    v16bf r;
    cvt_run(r, 0, base + kb);
    cvt_run(r, 8, base + kb + 16);
    return r;
}

// Fragment from a K=48 f32 weight row, zero-padded to K=64 (kt in {0,1}).
__device__ __forceinline__ v16bf gfrag48(const float* base, const int kt, const int hs) {
    v16bf r;
    const int kb = kt * 32 + (hs << 3);
    cvt_run(r, 0, base + kb);            // k in [kb, kb+8) < 48 always
    if (kt == 0) {
        cvt_run(r, 8, base + kb + 16);   // k in [16,32) valid
    } else {
        #pragma unroll
        for (int i = 8; i < 16; ++i) r[i] = (__bf16)0.0f;   // k >= 48 -> 0
    }
    return r;
}

// Fragment from an LDS bf16 row (row stride 64 halves): two ds_load_b128.
__device__ __forceinline__ v16bf lfrag(const __bf16* rowp, const int kb) {
    union { uint4 q[2]; v16bf v; } u;
    u.q[0] = *reinterpret_cast<const uint4*>(rowp + kb);
    u.q[1] = *reinterpret_cast<const uint4*>(rowp + kb + 16);
    return u.v;
}

// ---------------------------------------------------------------------------
// Fused agg path: proj -> qkv -> 5x5x2-head attention -> out proj.
// One workgroup = 5 waves (160 threads) handles 80 consecutive rows j
// (= 5 row-tiles of 16 = 16 attention groups). Wave t owns row-tile t.
// ---------------------------------------------------------------------------
__global__ __launch_bounds__(160)
void agg_attn_kernel(const float* __restrict__ x,
                     const float* __restrict__ agg_W,   // (5,48,32)
                     const float* __restrict__ agg_b,   // (5,48)
                     const float* __restrict__ in_w,    // (144,48)
                     const float* __restrict__ in_b,    // (144,)
                     const float* __restrict__ out_w,   // (48,48)
                     const float* __restrict__ out_b,   // (48,)
                     float* __restrict__ out)           // (B,N,512)
{
    __shared__ __bf16 Sbuf[80][64];    // S then reused for O; cols 48..63 = 0
    __shared__ float  QKV[80][144];    // q:0..47 k:48..95 v:96..143 (fp32)

    const int tid  = threadIdx.x;
    const int lane = tid & 31;
    const int t    = tid >> 5;         // wave == row tile, 0..4
    const int col  = lane & 15;
    const int hs   = (lane >> 4) & 1;

    const long j0 = (long)blockIdx.x * 80;

    // zero the K-padding columns (48..63) once; they stay zero across S/O reuse
    for (int r = tid; r < 80; r += 160) {
        *reinterpret_cast<uint4*>(&Sbuf[r][48]) = uint4{0u, 0u, 0u, 0u};
        *reinterpret_cast<uint4*>(&Sbuf[r][56]) = uint4{0u, 0u, 0u, 0u};
    }

    // ---- Stage A: S[j][e] = x_agg[j] @ agg_W[a]^T + agg_b[a]  (K=32) ----
    {
        const long jt = j0 + (long)t * 16;          // tile-aligned: same 'a'
        const int  a  = (int)((jt % 20480) / 4096);
        const long j  = jt + col;                   // this lane's A row
        const long xr = (j / 20480) * 4096 + (j % 4096);

        const v16bf af = gfrag(x + xr * 256 + a * 32, hs << 3);
        #pragma unroll
        for (int et = 0; et < 3; ++et) {
            const int e = et * 16 + col;
            const v16bf bf = gfrag(agg_W + ((long)(a * 48 + e)) * 32, hs << 3);
            v8f c = {};
            c = __builtin_amdgcn_wmma_f32_16x16x32_bf16(false, af, false, bf,
                                                        (short)0, c, false, false);
            const float bias = agg_b[a * 48 + e];
            #pragma unroll
            for (int v = 0; v < 8; ++v) {
                const int m = v + (hs << 3);
                Sbuf[t * 16 + m][e] = (__bf16)(c[v] + bias);
            }
        }
    }
    __syncthreads();

    // ---- Stage B: QKV = S @ in_w^T + in_b  (K=48 padded to 64, N=144) ----
    {
        const __bf16* srow = &Sbuf[t * 16 + col][0];
        v16bf af[2];
        #pragma unroll
        for (int kt = 0; kt < 2; ++kt) af[kt] = lfrag(srow, kt * 32 + (hs << 3));

        for (int nt = 0; nt < 9; ++nt) {
            const int nq = nt * 16 + col;
            v8f c = {};
            #pragma unroll
            for (int kt = 0; kt < 2; ++kt) {
                const v16bf bf = gfrag48(in_w + nq * 48, kt, hs);
                c = __builtin_amdgcn_wmma_f32_16x16x32_bf16(false, af[kt], false, bf,
                                                            (short)0, c, false, false);
            }
            const float bias = in_b[nq];
            #pragma unroll
            for (int v = 0; v < 8; ++v) {
                const int m = v + (hs << 3);
                QKV[t * 16 + m][nq] = c[v] + bias;
            }
        }
    }
    __syncthreads();

    // ---- Stage C: 5x5 softmax attention; 160 tasks = (16 g, 2 h, 5 qi) ----
    {
        const int g  = tid / 10;
        const int rm = tid - g * 10;
        const int h  = rm / 5;
        const int qi = rm - h * 5;
        const int r0 = g * 5;
        const int qo = h * 24, ko = 48 + h * 24, vo = 96 + h * 24;
        const float scale = 0.20412414523193154f;   // 1/sqrt(24)

        float q[24];
        #pragma unroll
        for (int d = 0; d < 24; ++d) q[d] = QKV[r0 + qi][qo + d];

        float s[5];
        #pragma unroll
        for (int ki = 0; ki < 5; ++ki) {
            float acc = 0.f;
            #pragma unroll
            for (int d = 0; d < 24; ++d) acc += q[d] * QKV[r0 + ki][ko + d];
            s[ki] = acc * scale;
        }
        const float mx = fmaxf(fmaxf(fmaxf(s[0], s[1]), fmaxf(s[2], s[3])), s[4]);
        float sum = 0.f;
        #pragma unroll
        for (int ki = 0; ki < 5; ++ki) { s[ki] = __expf(s[ki] - mx); sum += s[ki]; }
        const float inv = 1.f / sum;
        #pragma unroll
        for (int ki = 0; ki < 5; ++ki) s[ki] *= inv;

        #pragma unroll
        for (int d = 0; d < 24; d += 2) {
            float o0 = 0.f, o1 = 0.f;
            #pragma unroll
            for (int ki = 0; ki < 5; ++ki) {
                o0 += s[ki] * QKV[r0 + ki][vo + d];
                o1 += s[ki] * QKV[r0 + ki][vo + d + 1];
            }
            union { __bf16 h2[2]; unsigned u; } pk;
            pk.h2[0] = (__bf16)o0; pk.h2[1] = (__bf16)o1;
            *reinterpret_cast<unsigned*>(&Sbuf[r0 + qi][qo + d]) = pk.u;  // O
        }
    }
    __syncthreads();

    // ---- Stage D: AO = O @ out_w^T + out_b; scatter to agg output block ----
    {
        const __bf16* orow = &Sbuf[t * 16 + col][0];
        v16bf af[2];
        #pragma unroll
        for (int kt = 0; kt < 2; ++kt) af[kt] = lfrag(orow, kt * 32 + (hs << 3));

        #pragma unroll
        for (int nt = 0; nt < 3; ++nt) {
            const int n = nt * 16 + col;
            v8f c = {};
            #pragma unroll
            for (int kt = 0; kt < 2; ++kt) {
                const v16bf bf = gfrag48(out_w + n * 48, kt, hs);
                c = __builtin_amdgcn_wmma_f32_16x16x32_bf16(false, af[kt], false, bf,
                                                            (short)0, c, false, false);
            }
            const float bias = out_b[n];
            #pragma unroll
            for (int v = 0; v < 8; ++v) {
                const int m  = v + (hs << 3);
                const long j = j0 + t * 16 + m;
                const long r = j / 5;
                const int a2 = (int)(j - r * 5);
                out[r * 512 + a2 * 48 + n] = c[v] + bias;
            }
        }
    }
}

// ---------------------------------------------------------------------------
// Extra path: (B*N, 96) @ (96, 272)^T + b -> out[..., 240:512].
// K = 96 = 3 exact bf16-WMMA K-slices; N = 272 = 17 tiles. 4 waves/WG,
// one 16-row M-tile per wave (64 rows per workgroup).
// ---------------------------------------------------------------------------
__global__ __launch_bounds__(128)
void extra_kernel(const float* __restrict__ x,
                  const float* __restrict__ ew,   // (272,96)
                  const float* __restrict__ eb,   // (272,)
                  float* __restrict__ out)
{
    const int tid  = threadIdx.x;
    const int lane = tid & 31;
    const int wave = tid >> 5;
    const int col  = lane & 15;
    const int hs   = (lane >> 4) & 1;

    const long row0 = (long)blockIdx.x * 64 + wave * 16;
    const float* px = x + (row0 + col) * 256 + 160;   // extra_feat slice

    v16bf af[3];
    #pragma unroll
    for (int kt = 0; kt < 3; ++kt) af[kt] = gfrag(px, kt * 32 + (hs << 3));

    for (int nt = 0; nt < 17; ++nt) {
        const int n = nt * 16 + col;
        v8f c = {};
        #pragma unroll
        for (int kt = 0; kt < 3; ++kt) {
            const v16bf bf = gfrag(ew + (long)n * 96, kt * 32 + (hs << 3));
            c = __builtin_amdgcn_wmma_f32_16x16x32_bf16(false, af[kt], false, bf,
                                                        (short)0, c, false, false);
        }
        const float bias = eb[n];
        #pragma unroll
        for (int v = 0; v < 8; ++v) {
            const int m = v + (hs << 3);
            out[(row0 + m) * 512 + 240 + n] = c[v] + bias;
        }
    }
}

extern "C" void kernel_launch(void* const* d_in, const int* in_sizes, int n_in,
                              void* d_out, int out_size, void* d_ws, size_t ws_size,
                              hipStream_t stream) {
    const float* x    = (const float*)d_in[0];
    const float* aggW = (const float*)d_in[1];
    const float* aggb = (const float*)d_in[2];
    const float* inw  = (const float*)d_in[3];
    const float* inb  = (const float*)d_in[4];
    const float* outw = (const float*)d_in[5];
    const float* outb = (const float*)d_in[6];
    const float* exw  = (const float*)d_in[7];
    const float* exb  = (const float*)d_in[8];
    float* out = (float*)d_out;

    // 32*4096*5 = 655360 rows j; 80 rows per WG -> 8192 WGs of 5 waves.
    agg_attn_kernel<<<8192, 160, 0, stream>>>(x, aggW, aggb, inw, inb, outw, outb, out);
    // 32*4096 = 131072 rows; 64 rows per WG -> 2048 WGs of 4 waves.
    extra_kernel<<<2048, 128, 0, stream>>>(x, exw, exb, out);
}
